// NeuralSheet_46102178955921
// MI455X (gfx1250) — matrix operand order repaired and checked by reading, not compiled, not run on hardware
//
#include <hip/hip_runtime.h>
#include <hip/hip_bf16.h>
#include <math.h>

// MI455X / gfx1250 implementation.
//
// Roofline: the 16 recurrent iterations are 6400x6400 matvecs (bandwidth
// bound, ~0.5 flop/byte). Only two distinct effective matrices exist
// (the (t,u) schedule has 2 states), so we prebuild
//   A1 = Wexc - 1.6*Winh              (iterations 0..4)
//   A2 = Wexc - 0.4*lri - 0.6*Winh    (iterations 5..15)
// in bf16: 2 x 81.9 MB = 163.8 MB, which fits the 192 MB L2, so all 16
// matvec sweeps stream from L2 (~30 us total HBM cost for the one-time
// fp32 prep). The matvec uses v_wmma_f32_16x16x32_bf16 with the r vector
// replicated across B's 16 columns; compute waste is irrelevant when
// A-traffic-bound. Read-once fp32 inputs use non-temporal loads so they
// don't evict A1/A2 from L2.
//
// Workspace layout (needs ~164 MB):
//   [0)                 A1  bf16  N*N
//   [N*N*2)             A2  bf16  N*N
//   [N*N*4)             afferent f32 N
//   [N*N*4 + 4N)        r ping  bf16 N
//   [... + 2N)          r pong  bf16 N

typedef __attribute__((ext_vector_type(16))) __bf16 v16bf;
typedef __attribute__((ext_vector_type(8)))  __bf16 v8bf;
typedef __attribute__((ext_vector_type(8)))  float  v8f;

#define SHEET 80
#define NN    6400      // SHEET*SHEET
#define KINP  1024      // 32*32

// ---------------------------------------------------------------------------
// Build A1/A2 (one block per row; inh+corr rows cached in LDS: 51.2 KB).
// All fp32 matrix reads are non-temporal: read-once, keep L2 for A1/A2.
// ---------------------------------------------------------------------------
__global__ __launch_bounds__(256)
void prep_rows(const float* __restrict__ Wexc,
               const float* __restrict__ Winh,
               const float* __restrict__ Wcorr,
               __bf16* __restrict__ A1,
               __bf16* __restrict__ A2)
{
    __shared__ float sInh[NN];
    __shared__ float sCor[NN];
    __shared__ float red[256];

    const int n = blockIdx.x;
    const int t = threadIdx.x;
    const size_t base = (size_t)n * NN;

    for (int i = t; i < NN; i += 256) {
        sInh[i] = __builtin_nontemporal_load(Winh + base + i);
        sCor[i] = __builtin_nontemporal_load(Wcorr + base + i);
    }
    __syncthreads();

    // row max of untuned_inh
    float m = -3.4e38f;
    for (int i = t; i < NN; i += 256) m = fmaxf(m, sInh[i]);
    red[t] = m;
    __syncthreads();
    for (int s = 128; s > 0; s >>= 1) {
        if (t < s) red[t] = fmaxf(red[t], red[t + s]);
        __syncthreads();
    }
    const float permax = red[0];
    __syncthreads();

    // sum of corr * (1 - inh/permax)
    float acc = 0.0f;
    const float inv_pm = 1.0f / permax;
    for (int i = t; i < NN; i += 256)
        acc += sCor[i] * (1.0f - sInh[i] * inv_pm);
    red[t] = acc;
    __syncthreads();
    for (int s = 128; s > 0; s >>= 1) {
        if (t < s) red[t] += red[t + s];
        __syncthreads();
    }
    const float inv_sum = 1.0f / (red[0] + 1e-11f);

    // A1 = exc - 1.6*inh ; A2 = exc - 0.4*lri - 0.6*inh
    for (int i = t; i < NN; i += 256) {
        float inh = sInh[i];
        float lri = sCor[i] * (1.0f - inh * inv_pm) * inv_sum;
        float exc = __builtin_nontemporal_load(Wexc + base + i);
        A1[base + i] = (__bf16)(exc - 1.6f * inh);
        A2[base + i] = (__bf16)(exc - 0.4f * lri - 0.6f * inh);
    }
}

// ---------------------------------------------------------------------------
// afferent = W_aff(6400x1024) @ input ; one wave per row, input in LDS.
// W_aff is read-once fp32 -> non-temporal.
// ---------------------------------------------------------------------------
__global__ __launch_bounds__(256)
void afferent_matvec(const float* __restrict__ W,
                     const float* __restrict__ x,
                     float* __restrict__ y)
{
    __shared__ float sx[KINP];
    for (int i = threadIdx.x; i < KINP; i += 256) sx[i] = x[i];
    __syncthreads();

    const int wave = threadIdx.x >> 5;
    const int lane = threadIdx.x & 31;
    const int row  = blockIdx.x * 8 + wave;
    const float* wr = W + (size_t)row * KINP;

    float acc = 0.0f;
    for (int k = lane; k < KINP; k += 32)
        acc += __builtin_nontemporal_load(wr + k) * sx[k];
    for (int off = 16; off > 0; off >>= 1) acc += __shfl_xor(acc, off, 32);
    if (lane == 0) y[row] = acc;
}

__global__ __launch_bounds__(256)
void zero_r(__bf16* __restrict__ r)
{
    for (int i = threadIdx.x; i < NN; i += 256) r[i] = (__bf16)0.0f;
}

// ---------------------------------------------------------------------------
// One recurrent iteration: y = A @ r ; r' = tanh(relu(aff + 2y - thr))
// Block = 128 threads = 4 waves; block owns a 16-row stripe; waves split K.
// Each wave does 50 v_wmma_f32_16x16x32_bf16 accumulations streaming A from
// L2 (default RT temporal hint keeps A1/A2 resident).
// ---------------------------------------------------------------------------
__global__ __launch_bounds__(128)
void sheet_iter(const __bf16* __restrict__ A,
                const __bf16* __restrict__ r_in,
                const float*  __restrict__ aff,
                const float*  __restrict__ thr,
                __bf16* __restrict__ r_out,
                float*  __restrict__ out_f32)   // non-null on last iteration
{
    __shared__ float red[4][16];

    const int wave    = threadIdx.x >> 5;
    const int lane    = threadIdx.x & 31;
    const int m       = lane & 15;      // row within stripe
    const int hi      = lane >> 4;      // half-wave select
    const int rowBase = blockIdx.x * 16;

    const __bf16* Arow = A + (size_t)(rowBase + m) * NN;

    const int CHUNKS   = NN / 32;       // 200 K-chunks of 32
    const int PER_WAVE = CHUNKS / 4;    // 50 per wave
    const int c0 = wave * PER_WAVE;

    v8f acc = {};
    for (int c = 0; c < PER_WAVE; ++c) {
        const int kb = (c0 + c) * 32;

        // A fragment 16x32 bf16: lane holds K=[ka..ka+7] and K=[ka+16..ka+23]
        const int ka = kb + hi * 8;
        v8bf alo = *(const v8bf*)(Arow + ka);
        v8bf ahi = *(const v8bf*)(Arow + ka + 16);
        v16bf a;
#pragma unroll
        for (int i = 0; i < 8; ++i) { a[i] = alo[i]; a[i + 8] = ahi[i]; }

        // B fragment 32x16 bf16: vector r replicated across all 16 columns.
        // lane holds K=[ka..ka+7] and K=[ka+16..ka+23], matching A.
        v8bf blo = *(const v8bf*)(r_in + ka);
        v8bf bhi = *(const v8bf*)(r_in + ka + 16);
        v16bf b;
#pragma unroll
        for (int i = 0; i < 8; ++i) { b[i] = blo[i]; b[i + 8] = bhi[i]; }

        acc = __builtin_amdgcn_wmma_f32_16x16x32_bf16(
            /*neg_a=*/false, a, /*neg_b=*/false, b,
            /*c_mod=*/(short)0, acc, /*reuse_a=*/false, /*reuse_b=*/false);
    }

    // Column N=0 of D: lane 0 holds rows 0..7 (v0..v7), lane 16 rows 8..15.
    if (lane == 0) {
#pragma unroll
        for (int v = 0; v < 8; ++v) red[wave][v] = acc[v];
    }
    if (lane == 16) {
#pragma unroll
        for (int v = 0; v < 8; ++v) red[wave][8 + v] = acc[v];
    }
    __syncthreads();

    if (threadIdx.x < 16) {
        const int t = threadIdx.x;
        const float y = red[0][t] + red[1][t] + red[2][t] + red[3][t];
        const int row = rowBase + t;
        const float z  = aff[row] + 2.0f * y - thr[row];
        const float rv = tanhf(fmaxf(z, 0.0f));
        r_out[row] = (__bf16)rv;
        if (out_f32) out_f32[row] = rv;
    }
}

// ---------------------------------------------------------------------------
extern "C" void kernel_launch(void* const* d_in, const int* in_sizes, int n_in,
                              void* d_out, int out_size, void* d_ws, size_t ws_size,
                              hipStream_t stream)
{
    const float* input_crop = (const float*)d_in[0];   // 1024
    const float* aff_w      = (const float*)d_in[1];   // 6400x1024
    const float* w_exc      = (const float*)d_in[2];   // 6400x6400
    const float* w_inh      = (const float*)d_in[3];   // 6400x6400
    const float* w_corr     = (const float*)d_in[4];   // 6400x6400
    const float* thr        = (const float*)d_in[5];   // 6400

    char* ws = (char*)d_ws;
    __bf16* A1  = (__bf16*)ws;
    __bf16* A2  = (__bf16*)(ws + (size_t)NN * NN * 2);
    float*  aff = (float*) (ws + (size_t)NN * NN * 4);
    __bf16* rA  = (__bf16*)(ws + (size_t)NN * NN * 4 + (size_t)NN * 4);
    __bf16* rB  = rA + NN;

    prep_rows<<<NN, 256, 0, stream>>>(w_exc, w_inh, w_corr, A1, A2);
    afferent_matvec<<<NN / 8, 256, 0, stream>>>(aff_w, input_crop, aff);
    zero_r<<<1, 256, 0, stream>>>(rA);

    __bf16* rin  = rA;
    __bf16* rout = rB;
    for (int it = 0; it < 16; ++it) {
        const __bf16* A = (it < 5) ? A1 : A2;   // two-state (t,u) schedule
        float* outp = (it == 15) ? (float*)d_out : (float*)nullptr;
        sheet_iter<<<NN / 16, 128, 0, stream>>>(A, rin, aff, thr, rout, outp);
        __bf16* tmp = rin; rin = rout; rout = tmp;
    }
}
